// SpectreHead_7825430413853
// MI455X (gfx1250) — matrix operand order repaired and checked by reading, not compiled, not run on hardware
//
#include <hip/hip_runtime.h>
#include <hip/hip_bf16.h>

typedef __attribute__((ext_vector_type(16))) _Float16 v16h;
typedef __attribute__((ext_vector_type(8)))  float    v8f;
typedef unsigned int v4u_t __attribute__((ext_vector_type(4)));
typedef int          v8i_t __attribute__((ext_vector_type(8)));
typedef int          v4i_t __attribute__((ext_vector_type(4)));

#define B_     8
#define N_     4096
#define D_     1024
#define G_     4
#define DG_    256
#define NFFT_  4096
#define FHALF_ 2049
#define NBUCK_ 45
#define DGATE_ 256
#define BW_    4
#define KTAP_  9

typedef union { v16h v; _Float16 h[16]; unsigned u[8]; } frag16;
typedef union { _Float16 h[4]; unsigned long long u64; } pack4h;

// ---------------------------------------------------------------------------
// Kernel 1: xbar[b][d] = mean_n x[b][n][d]   (partial sums + atomicAdd)
// ---------------------------------------------------------------------------
__global__ __launch_bounds__(256) void mean_kernel(const float* __restrict__ x,
                                                   float* __restrict__ xbar) {
  int d  = blockIdx.x * 256 + threadIdx.x;
  int b  = blockIdx.z;
  int n0 = blockIdx.y * 128;
  const float* p = x + ((size_t)b * N_ + n0) * D_ + d;
  float s = 0.f;
#pragma unroll 4
  for (int n = 0; n < 128; ++n) s += p[(size_t)n * D_];
  atomicAdd(&xbar[b * D_ + d], s * (1.0f / (float)N_));
}

// ---------------------------------------------------------------------------
// Kernel 2: per-batch gate pipeline -> gate_half (complex f32) in ws
// ---------------------------------------------------------------------------
__global__ __launch_bounds__(256) void gate_kernel(
    const float* __restrict__ xbar, const float* __restrict__ Wq,
    const float* __restrict__ gw1, const float* __restrict__ gb1,
    const float* __restrict__ gw2, const float* __restrict__ gb2,
    const float* __restrict__ ln_g, const float* __restrict__ ln_b,
    const float* __restrict__ mod_bias,
    const float* __restrict__ tk_r, const float* __restrict__ tk_i,
    float* __restrict__ ghalf /* [B][G][FHALF][2] */) {
  __shared__ float s_x[D_];
  __shared__ float s_q[D_];
  __shared__ float s_h[DGATE_];
  __shared__ float s_red[256];
  __shared__ float s_red2[256];
  __shared__ float s_ga[2][G_ * NBUCK_ * 2];

  int b = blockIdx.x, t = threadIdx.x;
  for (int i = t; i < D_; i += 256) s_x[i] = xbar[b * D_ + i];
  __syncthreads();

  for (int d = t; d < D_; d += 256) {
    const float* w = Wq + (size_t)d * D_;
    float s = 0.f;
    for (int j = 0; j < D_; ++j) s = fmaf(s_x[j], w[j], s);
    s_q[d] = s;
  }
  __syncthreads();

  float ls = 0.f, ls2 = 0.f;
  for (int d = t; d < D_; d += 256) { float v = s_q[d]; ls += v; ls2 += v * v; }
  s_red[t] = ls; s_red2[t] = ls2;
  __syncthreads();
  for (int off = 128; off > 0; off >>= 1) {
    if (t < off) { s_red[t] += s_red[t + off]; s_red2[t] += s_red2[t + off]; }
    __syncthreads();
  }
  float mu  = s_red[0]  * (1.f / (float)D_);
  float var = s_red2[0] * (1.f / (float)D_) - mu * mu;
  float inv = rsqrtf(var + 1e-5f);
  __syncthreads();
  for (int d = t; d < D_; d += 256)
    s_q[d] = (s_q[d] - mu) * inv * ln_g[d] + ln_b[d];
  __syncthreads();

  {
    const float* w = gw1 + (size_t)t * D_;
    float s = gb1[t];
    for (int j = 0; j < D_; ++j) s = fmaf(s_q[j], w[j], s);
    s_h[t] = 0.5f * s * (1.f + erff(s * 0.70710678118654752f));
  }
  __syncthreads();

  for (int o = t; o < G_ * NBUCK_ * 2; o += 256) {
    const float* w = gw2 + (size_t)o * DGATE_;
    float s = gb2[o];
    for (int j = 0; j < DGATE_; ++j) s = fmaf(s_h[j], w[j], s);
    s_ga[0][o] = s;
  }
  __syncthreads();

  for (int idx = t; idx < G_ * NBUCK_; idx += 256) {
    int g = idx / NBUCK_, u = idx % NBUCK_;
    float cr = 0.f, ci = 0.f;
    for (int k = 0; k < KTAP_; ++k) {
      int uu = u - BW_ + k;
      uu = (uu % NBUCK_ + NBUCK_) % NBUCK_;
      float ar = s_ga[0][(g * NBUCK_ + uu) * 2 + 0];
      float ai = s_ga[0][(g * NBUCK_ + uu) * 2 + 1];
      float kr = tk_r[k], ki = tk_i[k];
      cr += kr * ar - ki * ai;
      ci += kr * ai + ki * ar;
    }
    s_ga[1][(g * NBUCK_ + u) * 2 + 0] = s_ga[0][(g * NBUCK_ + u) * 2 + 0] + cr;
    s_ga[1][(g * NBUCK_ + u) * 2 + 1] = s_ga[0][(g * NBUCK_ + u) * 2 + 1] + ci;
  }
  __syncthreads();

  for (int f = t; f < FHALF_; f += 256) {
    float p  = (float)f * (44.0f / 2048.0f);
    float fi = floorf(p);
    float tt = p - fi;
    int   i0 = (int)fi;
    const float A = -0.75f;
    float tp1 = tt + 1.f, tm1 = 1.f - tt, tm2 = 2.f - tt;
    float w0 = A * tp1 * tp1 * tp1 - 5.f * A * tp1 * tp1 + 8.f * A * tp1 - 4.f * A;
    float w1 = (A + 2.f) * tt * tt * tt - (A + 3.f) * tt * tt + 1.f;
    float w2 = (A + 2.f) * tm1 * tm1 * tm1 - (A + 3.f) * tm1 * tm1 + 1.f;
    float w3 = A * tm2 * tm2 * tm2 - 5.f * A * tm2 * tm2 + 8.f * A * tm2 - 4.f * A;
    int ja = min(max(i0 - 1, 0), 44), jb = min(max(i0, 0), 44);
    int jc = min(max(i0 + 1, 0), 44), jd = min(max(i0 + 2, 0), 44);
    for (int g = 0; g < G_; ++g) {
      const float* ga = &s_ga[1][g * NBUCK_ * 2];
      float zr = w0 * ga[ja * 2] + w1 * ga[jb * 2] + w2 * ga[jc * 2] + w3 * ga[jd * 2];
      float zi = w0 * ga[ja * 2 + 1] + w1 * ga[jb * 2 + 1] + w2 * ga[jc * 2 + 1] + w3 * ga[jd * 2 + 1];
      float mag = sqrtf(zr * zr + zi * zi);
      float sc  = fmaxf(mag + mod_bias[g * FHALF_ + f], 0.f) * rsqrtf(mag * mag + 1e-8f);
      size_t o = (((size_t)b * G_ + g) * FHALF_ + f) * 2;
      ghalf[o]     = zr * sc;
      ghalf[o + 1] = zi * sc;
    }
  }
}

// ---------------------------------------------------------------------------
// Kernel 3: k2[bg][n] = irfft(gate_half[bg])[n], duplicated to length 8192
// ---------------------------------------------------------------------------
__global__ __launch_bounds__(256) void ikern_kernel(const float* __restrict__ ghalf,
                                                    _Float16* __restrict__ k2) {
  __shared__ float s_re[FHALF_];
  __shared__ float s_im[FHALF_];
  int bg = blockIdx.y;
  int n  = blockIdx.x * 256 + threadIdx.x;
  const float* gh = ghalf + (size_t)bg * FHALF_ * 2;
  for (int f = threadIdx.x; f < FHALF_; f += 256) {
    s_re[f] = gh[f * 2];
    s_im[f] = gh[f * 2 + 1];
  }
  __syncthreads();

  float omega = 6.28318530717958647692f * (float)n / 4096.0f;
  float cA, sA;
  sincosf(omega, &sA, &cA);
  float sum = s_re[0] + ((n & 1) ? -s_re[2048] : s_re[2048]);
  float acc = 0.f;
  int f = 1;
  while (f < 2048) {
    float s, c;
    sincosf(omega * (float)f, &s, &c);
    int fe = min(f + 128, 2048);
    for (; f < fe; ++f) {
      acc = fmaf(s_re[f], c, acc);
      acc = fmaf(-s_im[f], s, acc);
      float cn = c * cA - s * sA;
      float sn = s * cA + c * sA;
      c = cn; s = sn;
    }
  }
  float kv = (sum + 2.f * acc) * (1.0f / 4096.0f);
  size_t base = (size_t)bg * 8192;
  k2[base + n]        = (_Float16)kv;
  k2[base + n + 4096] = (_Float16)kv;
}

// ---------------------------------------------------------------------------
// Kernel 4: V = x @ Wv^T in f16. WMMA, K-step 64, float4 staging loads.
// ---------------------------------------------------------------------------
__global__ __launch_bounds__(256) void vgemm_kernel(const float* __restrict__ x,
                                                    const float* __restrict__ Wv,
                                                    _Float16* __restrict__ Vh) {
  __shared__ __align__(16) _Float16 As[128][68];  // [m][k], pad 4 -> 136B rows
  __shared__ __align__(16) _Float16 Bs[128][68];  // [n][k]  (B[k][n] = Wv[n][k])
  int m0 = blockIdx.y * 128;
  int n0 = blockIdx.x * 128;
  int t    = threadIdx.x;
  int lane = t & 31, wid = t >> 5;
  int wm = (wid & 1) * 64;
  int wn = (wid >> 1) * 32;
  int l15 = lane & 15, lhi = lane >> 4;

  v8f acc[4][2];
#pragma unroll
  for (int mt = 0; mt < 4; ++mt)
#pragma unroll
    for (int nt = 0; nt < 2; ++nt) { v8f z = {}; acc[mt][nt] = z; }

  for (int k0 = 0; k0 < 1024; k0 += 64) {
    __syncthreads();
#pragma unroll
    for (int i = 0; i < 8; ++i) {           // 128x64 f32 -> f16, b128 loads
      int slot = t + i * 256;               // 0..2047 float4 slots
      int r = slot >> 4, c4 = (slot & 15) << 2;
      float4 ax = *(const float4*)&x[((size_t)(m0 + r)) * 1024 + k0 + c4];
      float4 bx = *(const float4*)&Wv[((size_t)(n0 + r)) * 1024 + k0 + c4];
      pack4h pa, pb;
      pa.h[0] = (_Float16)ax.x; pa.h[1] = (_Float16)ax.y;
      pa.h[2] = (_Float16)ax.z; pa.h[3] = (_Float16)ax.w;
      pb.h[0] = (_Float16)bx.x; pb.h[1] = (_Float16)bx.y;
      pb.h[2] = (_Float16)bx.z; pb.h[3] = (_Float16)bx.w;
      *(unsigned long long*)&As[r][c4] = pa.u64;
      *(unsigned long long*)&Bs[r][c4] = pb.u64;
    }
    if (k0 + 64 < 1024) {  // gfx1250 global_prefetch_b8 of next tiles
      __builtin_prefetch(&x[((size_t)(m0 + (t >> 1))) * 1024 + k0 + 64], 0, 1);
      __builtin_prefetch(&Wv[((size_t)(n0 + (t >> 1))) * 1024 + k0 + 64], 0, 1);
    }
    __syncthreads();

#pragma unroll
    for (int kk = 0; kk < 64; kk += 32) {
      frag16 afr[4], bfr[2];
#pragma unroll
      for (int mt = 0; mt < 4; ++mt) {
        int mi = wm + mt * 16 + l15;
#pragma unroll
        for (int e = 0; e < 16; ++e) {
          int K = (e & 7) + 8 * lhi + 16 * (e >> 3);
          afr[mt].h[e] = As[mi][kk + K];
        }
      }
#pragma unroll
      for (int nt = 0; nt < 2; ++nt) {
        int nn = wn + nt * 16 + l15;
        int kb = kk + lhi * 8;
#pragma unroll
        for (int e = 0; e < 16; ++e) bfr[nt].h[e] = Bs[nn][kb + (e & 7) + 16 * (e >> 3)];
      }
#pragma unroll
      for (int mt = 0; mt < 4; ++mt)
#pragma unroll
        for (int nt = 0; nt < 2; ++nt)
          acc[mt][nt] = __builtin_amdgcn_wmma_f32_16x16x32_f16(
              false, afr[mt].v, false, bfr[nt].v, (short)0, acc[mt][nt], false, false);
    }
  }

#pragma unroll
  for (int mt = 0; mt < 4; ++mt)
#pragma unroll
    for (int nt = 0; nt < 2; ++nt)
#pragma unroll
      for (int r = 0; r < 8; ++r) {
        int row = m0 + wm + mt * 16 + r + 8 * lhi;
        int col = n0 + wn + nt * 16 + l15;
        Vh[(size_t)row * 1024 + col] = (_Float16)acc[mt][nt][r];
      }
}

// ---------------------------------------------------------------------------
// Kernel 5: circulant (from k2) x V via WMMA.
//   V tiles fetched by the Tensor Data Mover into double-buffered LDS
//   (64 rows x 128 cols f16, LDS-padded 8 halves/row via D# pad fields),
//   overlapped with WMMA compute; synced via s_wait_tensorcnt + barrier.
// ---------------------------------------------------------------------------
__global__ __launch_bounds__(256) void circconv_kernel(const _Float16* __restrict__ Vh,
                                                       const _Float16* __restrict__ k2,
                                                       float* __restrict__ out) {
  __shared__ _Float16 kt[4224];                         // k2 slice: t = i0+1+v
  __shared__ __align__(16) _Float16 Vs[2][64][136];     // [buf][kj][n]
  int bg = blockIdx.z;
  int b = bg >> 2, g = bg & 3;
  int i0 = blockIdx.y * 128;
  int d0 = g * 256 + blockIdx.x * 128;
  int t    = threadIdx.x;
  int lane = t & 31, wid = t >> 5;
  int wm = (wid & 1) * 64, wn = (wid >> 1) * 32;
  int l15 = lane & 15, lhi = lane >> 4;

  const _Float16* vb = Vh + (size_t)b * N_ * D_ + d0;

  // TDM issue: one 2D tile load, 64 rows x 128 f16, row stride 1024 elems,
  // +16B LDS padding after every 256B row (pad_interval=64 DW, pad_amount=4 DW)
  auto tdm_issue = [&](int nb, int jj) {
    unsigned long long ga = (unsigned long long)(size_t)(vb + (size_t)jj * 1024);
    unsigned ldsOff = (unsigned)(size_t)(&Vs[nb][0][0]);
    v4u_t g0;
    g0.x = 1u;                                    // count=1, no gather/restore
    g0.y = ldsOff;                                // lds_addr
    g0.z = (unsigned)ga;                          // global_addr[31:0]
    g0.w = (unsigned)((ga >> 32) & 0x1FFFFFFull) | 0x80000000u;  // addr[56:32]|type=2
    v8i_t g1;
    g1[0] = (int)((1u << 16) | (1u << 20) | (5u << 22) | (3u << 25)); // 2B, pad on
    g1[1] = (int)(1024u << 16);                   // tensor_dim0 = 1024
    g1[2] = (int)(4096u << 16);                   // tensor_dim1 = 4096
    g1[3] = (int)(128u << 16);                    // tile_dim0 = 128
    g1[4] = 64;                                   // tile_dim1 = 64
    g1[5] = 1024;                                 // tensor_dim0_stride lo
    g1[6] = 0;
    g1[7] = 0;
    v4i_t g2 = {0, 0, 0, 0};
    v4i_t g3 = {0, 0, 0, 0};
#if __clang_major__ >= 23
    v8i_t g4 = {0, 0, 0, 0, 0, 0, 0, 0};
    __builtin_amdgcn_tensor_load_to_lds(g0, g1, g2, g3, g4, 0);
#else
    __builtin_amdgcn_tensor_load_to_lds(g0, g1, g2, g3, 0);
#endif
  };

  const _Float16* kg = k2 + (size_t)bg * 8192 + i0 + 1;
  for (int v = t; v < 4223; v += 256) kt[v] = kg[v];

  if (wid == 0) tdm_issue(0, 0);  // prologue: fill buffer 0

  v8f acc[4][2];
#pragma unroll
  for (int mt = 0; mt < 4; ++mt)
#pragma unroll
    for (int nt = 0; nt < 2; ++nt) { v8f z = {}; acc[mt][nt] = z; }

  int cur = 0;
  for (int j0 = 0; j0 < 4096; j0 += 64) {
    if (wid == 0) __builtin_amdgcn_s_wait_tensorcnt(0);  // current buffer ready
    __syncthreads();                                     // ...visible to all waves
    if (j0 + 64 < 4096 && wid == 0) tdm_issue(cur ^ 1, j0 + 64);  // overlap next

#pragma unroll
    for (int kk = 0; kk < 64; kk += 32) {
      frag16 afr[4], bfr[2];
#pragma unroll
      for (int mt = 0; mt < 4; ++mt) {
        int mi   = wm + mt * 16 + l15;
        int base = mi - (j0 + kk) + 4095;  // kt index for kj_local = 0
#pragma unroll
        for (int e = 0; e < 16; ++e) {
          int K = (e & 7) + 8 * lhi + 16 * (e >> 3);
          afr[mt].h[e] = kt[base - K];
        }
      }
#pragma unroll
      for (int nt = 0; nt < 2; ++nt) {
        int nn = wn + nt * 16 + l15;
        int kb = kk + lhi * 8;
#pragma unroll
        for (int e = 0; e < 16; ++e) bfr[nt].h[e] = Vs[cur][kb + (e & 7) + 16 * (e >> 3)][nn];
      }
#pragma unroll
      for (int mt = 0; mt < 4; ++mt)
#pragma unroll
        for (int nt = 0; nt < 2; ++nt)
          acc[mt][nt] = __builtin_amdgcn_wmma_f32_16x16x32_f16(
              false, afr[mt].v, false, bfr[nt].v, (short)0, acc[mt][nt], false, false);
    }
    cur ^= 1;
  }

#pragma unroll
  for (int mt = 0; mt < 4; ++mt)
#pragma unroll
    for (int nt = 0; nt < 2; ++nt)
#pragma unroll
      for (int r = 0; r < 8; ++r) {
        int row = i0 + wm + mt * 16 + r + 8 * lhi;
        int col = d0 + wn + nt * 16 + l15;
        out[((size_t)b * N_ + row) * D_ + col] = acc[mt][nt][r];
      }
}

// ---------------------------------------------------------------------------
extern "C" void kernel_launch(void* const* d_in, const int* in_sizes, int n_in,
                              void* d_out, int out_size, void* d_ws, size_t ws_size,
                              hipStream_t stream) {
  (void)in_sizes; (void)n_in; (void)out_size; (void)ws_size;
  const float* x        = (const float*)d_in[0];
  const float* Wq       = (const float*)d_in[1];
  const float* Wv       = (const float*)d_in[2];
  const float* gw1      = (const float*)d_in[3];
  const float* gb1      = (const float*)d_in[4];
  const float* gw2      = (const float*)d_in[5];
  const float* gb2      = (const float*)d_in[6];
  const float* ln_g     = (const float*)d_in[7];
  const float* ln_b     = (const float*)d_in[8];
  const float* mod_bias = (const float*)d_in[9];
  const float* tk_r     = (const float*)d_in[10];
  const float* tk_i     = (const float*)d_in[11];
  float* out = (float*)d_out;

  float*    ws    = (float*)d_ws;
  float*    xbar  = ws;                               // 8192 f32
  float*    ghalf = ws + 8192;                        // 131136 f32
  _Float16* k2    = (_Float16*)(ws + 8192 + 131136);  // 32*8192 f16
  _Float16* Vh    = k2 + (size_t)32 * 8192;           // 8*4096*1024 f16 (64 MB)

  hipMemsetAsync(xbar, 0, 8192 * sizeof(float), stream);
  mean_kernel<<<dim3(D_ / 256, 32, B_), 256, 0, stream>>>(x, xbar);
  gate_kernel<<<dim3(B_), 256, 0, stream>>>(xbar, Wq, gw1, gb1, gw2, gb2,
                                            ln_g, ln_b, mod_bias, tk_r, tk_i, ghalf);
  ikern_kernel<<<dim3(NFFT_ / 256, B_ * G_), 256, 0, stream>>>(ghalf, k2);
  vgemm_kernel<<<dim3(D_ / 128, (B_ * N_) / 128), 256, 0, stream>>>(x, Wv, Vh);
  circconv_kernel<<<dim3(2, N_ / 128, B_ * G_), 256, 0, stream>>>(Vh, k2, out);
}
